// GroupedQueryAttention_89300960018901
// MI455X (gfx1250) — matrix operand order repaired and checked by reading, not compiled, not run on hardware
//
#include <hip/hip_runtime.h>

typedef __bf16 bf16;
typedef __attribute__((ext_vector_type(16))) __bf16 v16bf;
typedef __attribute__((ext_vector_type(4)))  __bf16 v4bf;
typedef __attribute__((ext_vector_type(8)))  float  v8f;

#define SEQ      2048
#define BATCH    2
#define DMODEL   2048
#define KV_DIM   512
#define HEAD_DIM 128

union FragA { v16bf v; uint4 u[2]; };

__device__ __forceinline__ v8f wmma_bf16(v16bf a, v16bf b, v8f c) {
  // D = A(16x32 bf16) * B(32x16 bf16) + C(16x16 f32)
  return __builtin_amdgcn_wmma_f32_16x16x32_bf16(false, a, false, b, (short)0, c, false, false);
}

// CDNA5 async global->LDS copy (ASYNCcnt-tracked, no VGPR bounce).
// Per-lane: 16 bytes from gptr -> LDS at lptr. VDST = LDS byte offset VGPR.
__device__ __forceinline__ void async_copy_b128(const void* gptr, void* lptr) {
  unsigned lds_off = (unsigned)(size_t)lptr;              // low 32 bits = LDS offset
  unsigned long long ga = (unsigned long long)(size_t)gptr;
  asm volatile("global_load_async_to_lds_b128 %0, %1, off"
               :: "v"(lds_off), "v"(ga) : "memory");
}
__device__ __forceinline__ void wait_async0() {
  asm volatile("s_wait_asynccnt 0" ::: "memory");
}

// ---------------------------------------------------------------------------
// fp32 -> bf16 elementwise convert
// ---------------------------------------------------------------------------
__global__ void cvt_f32_bf16(const float* __restrict__ s, bf16* __restrict__ d, int n) {
  int i = blockIdx.x * blockDim.x + threadIdx.x;
  int stride = gridDim.x * blockDim.x;
  for (; i < n; i += stride) d[i] = (bf16)s[i];
}

// ---------------------------------------------------------------------------
// Tiled bf16 GEMM: C[M,N] = A[M,K] @ B[K,N]  (row-major bf16, f32 accum)
// OUT_F32_BIAS==0 -> bf16 out; ==1 -> f32 out + bias[col]
// Block: 256 thr = 8 waves; block tile 128(M) x 128(N); wave tile 64x32.
// Double-buffered LDS; A tiles via async-to-LDS, B tiles via reg prefetch
// (transposed scatter). One barrier per 32-deep K step.
// ---------------------------------------------------------------------------
template <int OUT_F32_BIAS>
__global__ __launch_bounds__(256) void gemm_bf16(
    const bf16* __restrict__ A, const bf16* __restrict__ B,
    const float* __restrict__ bias, bf16* __restrict__ Cb,
    float* __restrict__ Cf, int M, int N, int K) {
  __shared__ __align__(16) bf16 As[2][128][32];   // M x K-slice
  __shared__ __align__(16) bf16 BsT[2][128][32];  // N x K-slice (transposed)

  const int tid  = threadIdx.x;
  const int lane = tid & 31;
  const int w    = tid >> 5;
  const int wm   = w >> 2;        // 0..1  (64 rows each)
  const int wn   = w & 3;         // 0..3  (32 cols each)
  const int ln   = lane & 15;
  const int kh   = lane >> 4;
  const int m0   = blockIdx.y * 128;
  const int n0   = blockIdx.x * 128;

  // Async A staging: 512 b128 segments, 2 per thread
  const int arow0 = tid >> 2;              // segments tid, tid+256
  const int acv0  = (tid & 3) << 3;
  const int arow1 = (tid + 256) >> 2;
  const int acv1  = ((tid + 256) & 3) << 3;

  v8f acc[4][2] = {};

  const int ntiles = K >> 5;

  // Prologue: async A tile 0, B tile 0 into regs
  async_copy_b128(&A[(size_t)(m0 + arow0) * K + acv0], &As[0][arow0][acv0]);
  async_copy_b128(&A[(size_t)(m0 + arow1) * K + acv1], &As[0][arow1][acv1]);
  v4bf br[4];
#pragma unroll
  for (int t = 0; t < 4; ++t) {
    int slot = tid + t * 256;
    int krow = slot >> 5;
    int cv   = (slot & 31) << 2;
    br[t] = *(const v4bf*)(&B[(size_t)krow * N + n0 + cv]);
  }

  int cur = 0;
  for (int i = 0; i < ntiles; ++i) {
    const int kk = i << 5;
    // Scatter current B tile (transposed) into current buffer
#pragma unroll
    for (int t = 0; t < 4; ++t) {
      int slot = tid + t * 256;
      int krow = slot >> 5;
      int cv   = (slot & 31) << 2;
      BsT[cur][cv + 0][krow] = br[t][0];
      BsT[cur][cv + 1][krow] = br[t][1];
      BsT[cur][cv + 2][krow] = br[t][2];
      BsT[cur][cv + 3][krow] = br[t][3];
    }
    wait_async0();        // our async A segments for tile i are resident
    __syncthreads();      // tile i LDS complete WG-wide; prev-buffer readers done

    // Issue next tile's traffic now (overlaps with the 8 WMMAs below):
    if (i + 1 < ntiles) {
      const int kn = kk + 32;
      async_copy_b128(&A[(size_t)(m0 + arow0) * K + kn + acv0], &As[cur ^ 1][arow0][acv0]);
      async_copy_b128(&A[(size_t)(m0 + arow1) * K + kn + acv1], &As[cur ^ 1][arow1][acv1]);
#pragma unroll
      for (int t = 0; t < 4; ++t) {
        int slot = tid + t * 256;
        int krow = slot >> 5;
        int cv   = (slot & 31) << 2;
        br[t] = *(const v4bf*)(&B[(size_t)(kn + krow) * N + n0 + cv]);
      }
      if (i + 2 < ntiles) {  // L2 look-ahead (global_prefetch_b8)
        __builtin_prefetch(&A[(size_t)(m0 + arow0) * K + kn + 32 + acv0], 0, 0);
        __builtin_prefetch(&B[(size_t)(kn + 32 + (tid >> 3)) * N + n0 + ((tid & 7) << 4)], 0, 0);
      }
    }

    // Fragments + 8 WMMAs
    FragA af[4], bf_[2];
#pragma unroll
    for (int mt = 0; mt < 4; ++mt) {
      const bf16* ar = &As[cur][wm * 64 + mt * 16 + ln][0];
      af[mt].u[0] = *(const uint4*)(ar + kh * 16);       // K = kh*16 + 0..7
      af[mt].u[1] = *(const uint4*)(ar + kh * 16 + 8);   // K = kh*16 + 8..15
    }
#pragma unroll
    for (int nt = 0; nt < 2; ++nt) {
      const bf16* brr = &BsT[cur][wn * 32 + nt * 16 + ln][0];
      bf_[nt].u[0] = *(const uint4*)(brr + kh * 16);     // K = kh*16 + 0..7
      bf_[nt].u[1] = *(const uint4*)(brr + kh * 16 + 8); // K = kh*16 + 8..15
    }
#pragma unroll
    for (int mt = 0; mt < 4; ++mt)
#pragma unroll
      for (int nt = 0; nt < 2; ++nt)
        acc[mt][nt] = wmma_bf16(af[mt].v, bf_[nt].v, acc[mt][nt]);

    cur ^= 1;
  }

  // Epilogue: lane owns col n, rows r + 8*kh of each 16x16 tile
#pragma unroll
  for (int mt = 0; mt < 4; ++mt) {
#pragma unroll
    for (int nt = 0; nt < 2; ++nt) {
      int col = n0 + wn * 32 + nt * 16 + ln;
#pragma unroll
      for (int r = 0; r < 8; ++r) {
        int row = m0 + wm * 64 + mt * 16 + kh * 8 + r;
        float v = acc[mt][nt][r];
        if (OUT_F32_BIAS)
          Cf[(size_t)row * N + col] = v + bias[col];
        else
          Cb[(size_t)row * N + col] = (bf16)v;
      }
    }
  }
}

// ---------------------------------------------------------------------------
// Flash attention: grid = BATCH * 16 heads * (SEQ/128) q-blocks.
// Block = 8 waves; wave owns 16 q rows; K tiles of 32 with online softmax.
// K tile staged via async-to-LDS (rows contiguous); V transposed via VGPRs.
// ---------------------------------------------------------------------------
__global__ __launch_bounds__(256) void gqa_flash(
    const bf16* __restrict__ Q, const bf16* __restrict__ Kg,
    const bf16* __restrict__ Vg, bf16* __restrict__ Ctx) {
  __shared__ __align__(16) bf16 Ks[32][128];      // k-row x d
  __shared__ __align__(16) bf16 VsT[128][32];     // d x k-row (transposed)
  __shared__ __align__(16) bf16 Pt[8][16][32];    // per-wave P bounce

  const int tid  = threadIdx.x;
  const int lane = tid & 31;
  const int w    = tid >> 5;
  const int ln   = lane & 15;
  const int kh   = lane >> 4;

  const int wg   = blockIdx.x;
  const int qb   = wg & 15;          // q block (128 rows)
  const int head = (wg >> 4) & 15;   // head = g*4 + h (group-major)
  const int bi   = wg >> 8;          // batch
  const int grp  = head >> 2;

  const int qbase = qb * 128 + w * 16;
  const int hcol  = head * HEAD_DIM;
  const int kvcol = grp * HEAD_DIM;
  const float scale = 0.08838834764831845f;  // 1/sqrt(128)

  // Async K staging: 512 b128 segments (16 per 128-elem row), 2 per thread
  const int krow0 = tid >> 3;
  const int kcv0  = (tid & 7) << 3;        // slots tid (row 0..31 upper half?)
  const int krow1 = (tid + 256) >> 3;
  const int kcv1  = ((tid + 256) & 7) << 3;

  // Q fragments (A-matrix layout), one per 32-wide d chunk, kept in VGPRs
  FragA aq[4];
  {
    const bf16* qrow = &Q[(size_t)(bi * SEQ + qbase + ln) * DMODEL + hcol];
#pragma unroll
    for (int c = 0; c < 4; ++c) {
      aq[c].u[0] = *(const uint4*)(qrow + c * 32 + kh * 16);
      aq[c].u[1] = *(const uint4*)(qrow + c * 32 + kh * 16 + 8);
    }
  }

  v8f acc[8] = {};          // ctx 16x128, lane owns col j*16+ln, rows r+8*kh
  float m_r[8], l_r[8];
#pragma unroll
  for (int r = 0; r < 8; ++r) { m_r[r] = -3.0e38f; l_r[r] = 0.f; }

  const int nkb = qb * 4 + 4;  // causal bound for whole block
  for (int kb = 0; kb < nkb; ++kb) {
    const int kbase = kb * 32;
    // K tile: async global->LDS
    {
      size_t g0 = (size_t)(bi * SEQ + kbase + krow0) * KV_DIM + kvcol + kcv0;
      size_t g1 = (size_t)(bi * SEQ + kbase + krow1) * KV_DIM + kvcol + kcv1;
      async_copy_b128(&Kg[g0], &Ks[krow0][kcv0]);
      async_copy_b128(&Kg[g1], &Ks[krow1][kcv1]);
    }
    // V tile: transpose through VGPRs
#pragma unroll
    for (int t = 0; t < 4; ++t) {
      int slot = tid + t * 256;
      int krow = slot >> 5;
      int cv   = (slot & 31) << 2;
      v4bf vv = *(const v4bf*)(&Vg[(size_t)(bi * SEQ + kbase + krow) * KV_DIM + kvcol + cv]);
      VsT[cv + 0][krow] = vv[0];
      VsT[cv + 1][krow] = vv[1];
      VsT[cv + 2][krow] = vv[2];
      VsT[cv + 3][krow] = vv[3];
    }
    if (kb + 1 < nkb) {  // L2 look-ahead for next K/V tile
      __builtin_prefetch(&Kg[(size_t)(bi * SEQ + kbase + 32 + krow0) * KV_DIM + kvcol + kcv0], 0, 0);
      __builtin_prefetch(&Vg[(size_t)(bi * SEQ + kbase + 32 + krow0) * KV_DIM + kvcol + kcv0], 0, 0);
    }
    wait_async0();
    __syncthreads();

    if (kbase <= qbase + 15) {  // wave-uniform predicate (EXEC all-ones)
      // S = Q @ K^T : two 16x16 output tiles (k cols 0..15 and 16..31)
      v8f s0 = {}, s1 = {};
#pragma unroll
      for (int c = 0; c < 4; ++c) {
        FragA bk0, bk1;
        const bf16* kr0 = &Ks[ln][c * 32 + kh * 16];
        const bf16* kr1 = &Ks[16 + ln][c * 32 + kh * 16];
        bk0.u[0] = *(const uint4*)(kr0);
        bk0.u[1] = *(const uint4*)(kr0 + 8);
        bk1.u[0] = *(const uint4*)(kr1);
        bk1.u[1] = *(const uint4*)(kr1 + 8);
        s0 = wmma_bf16(aq[c].v, bk0.v, s0);
        s1 = wmma_bf16(aq[c].v, bk1.v, s1);
      }

      // Online softmax update (row reductions over 16 lanes sharing a row)
#pragma unroll
      for (int r = 0; r < 8; ++r) {
        int qg  = qbase + kh * 8 + r;
        int kg0 = kbase + ln;
        int kg1 = kbase + 16 + ln;
        float x0 = (kg0 <= qg) ? s0[r] * scale : -3.0e38f;
        float x1 = (kg1 <= qg) ? s1[r] * scale : -3.0e38f;
        float t = fmaxf(x0, x1);
        t = fmaxf(t, __shfl_xor(t, 1));
        t = fmaxf(t, __shfl_xor(t, 2));
        t = fmaxf(t, __shfl_xor(t, 4));
        t = fmaxf(t, __shfl_xor(t, 8));
        float mnew  = fmaxf(m_r[r], t);
        float alpha = __expf(m_r[r] - mnew);
        float p0 = (kg0 <= qg) ? __expf(x0 - mnew) : 0.f;
        float p1 = (kg1 <= qg) ? __expf(x1 - mnew) : 0.f;
        float rs = p0 + p1;
        rs += __shfl_xor(rs, 1);
        rs += __shfl_xor(rs, 2);
        rs += __shfl_xor(rs, 4);
        rs += __shfl_xor(rs, 8);
        l_r[r] = l_r[r] * alpha + rs;
        m_r[r] = mnew;
#pragma unroll
        for (int j = 0; j < 8; ++j) acc[j][r] *= alpha;
        Pt[w][kh * 8 + r][ln]      = (bf16)p0;
        Pt[w][kh * 8 + r][16 + ln] = (bf16)p1;
      }

      // P (D-layout) -> A-layout via per-wave LDS bounce, then ctx += P @ V
      FragA ap;
      const bf16* pr = &Pt[w][ln][0];
      ap.u[0] = *(const uint4*)(pr + kh * 16);
      ap.u[1] = *(const uint4*)(pr + kh * 16 + 8);
#pragma unroll
      for (int j = 0; j < 8; ++j) {
        FragA bv;
        const bf16* vr = &VsT[j * 16 + ln][kh * 16];
        bv.u[0] = *(const uint4*)(vr);
        bv.u[1] = *(const uint4*)(vr + 8);
        acc[j] = wmma_bf16(ap.v, bv.v, acc[j]);
      }
    }
    __syncthreads();
  }

  // Normalize and store ctx (bf16 for the output GEMM)
#pragma unroll
  for (int j = 0; j < 8; ++j) {
    int col = hcol + j * 16 + ln;
#pragma unroll
    for (int r = 0; r < 8; ++r) {
      int row = qbase + kh * 8 + r;
      float v = acc[j][r] / l_r[r];
      Ctx[(size_t)(bi * SEQ + row) * DMODEL + col] = (bf16)v;
    }
  }
}

// ---------------------------------------------------------------------------
extern "C" void kernel_launch(void* const* d_in, const int* in_sizes, int n_in,
                              void* d_out, int out_size, void* d_ws,
                              size_t ws_size, hipStream_t stream) {
  (void)in_sizes; (void)n_in; (void)out_size; (void)ws_size;
  const float* x  = (const float*)d_in[0];
  const float* Wq = (const float*)d_in[1];
  const float* Wk = (const float*)d_in[2];
  const float* Wv = (const float*)d_in[3];
  const float* Wo = (const float*)d_in[4];
  const float* bo = (const float*)d_in[5];
  float* out = (float*)d_out;

  char* ws = (char*)d_ws;
  size_t off = 0;
  auto alloc = [&](size_t bytes) {
    char* p = ws + off;
    off += (bytes + 255) & ~(size_t)255;
    return p;
  };
  const size_t NX = (size_t)BATCH * SEQ * DMODEL;   // 8388608
  bf16* xb  = (bf16*)alloc(NX * 2);
  bf16* wqb = (bf16*)alloc((size_t)DMODEL * DMODEL * 2);
  bf16* wkb = (bf16*)alloc((size_t)DMODEL * KV_DIM * 2);
  bf16* wvb = (bf16*)alloc((size_t)DMODEL * KV_DIM * 2);
  bf16* wob = (bf16*)alloc((size_t)DMODEL * DMODEL * 2);
  bf16* q_b = (bf16*)alloc(NX * 2);
  bf16* k_b = (bf16*)alloc((size_t)BATCH * SEQ * KV_DIM * 2);
  bf16* v_b = (bf16*)alloc((size_t)BATCH * SEQ * KV_DIM * 2);
  bf16* c_b = (bf16*)alloc(NX * 2);

  cvt_f32_bf16<<<2048, 256, 0, stream>>>(x,  xb,  (int)NX);
  cvt_f32_bf16<<<1024, 256, 0, stream>>>(Wq, wqb, DMODEL * DMODEL);
  cvt_f32_bf16<<<512,  256, 0, stream>>>(Wk, wkb, DMODEL * KV_DIM);
  cvt_f32_bf16<<<512,  256, 0, stream>>>(Wv, wvb, DMODEL * KV_DIM);
  cvt_f32_bf16<<<1024, 256, 0, stream>>>(Wo, wob, DMODEL * DMODEL);

  const int Mrows = BATCH * SEQ;  // 4096
  gemm_bf16<0><<<dim3(DMODEL / 128, Mrows / 128), 256, 0, stream>>>(
      xb, wqb, nullptr, q_b, nullptr, Mrows, DMODEL, DMODEL);
  gemm_bf16<0><<<dim3(KV_DIM / 128, Mrows / 128), 256, 0, stream>>>(
      xb, wkb, nullptr, k_b, nullptr, Mrows, KV_DIM, DMODEL);
  gemm_bf16<0><<<dim3(KV_DIM / 128, Mrows / 128), 256, 0, stream>>>(
      xb, wvb, nullptr, v_b, nullptr, Mrows, KV_DIM, DMODEL);

  gqa_flash<<<BATCH * 16 * (SEQ / 128), 256, 0, stream>>>(q_b, k_b, v_b, c_b);

  gemm_bf16<1><<<dim3(DMODEL / 128, Mrows / 128), 256, 0, stream>>>(
      c_b, wob, bo, nullptr, out, Mrows, DMODEL, DMODEL);
}
